// sLSTM_4999341932567
// MI455X (gfx1250) — compile-verified, matrix operand order checked by reading, not run on hardware
//
#include <hip/hip_runtime.h>

typedef _Float16 h16_t;
typedef __attribute__((ext_vector_type(16))) _Float16 v16h;
typedef __attribute__((ext_vector_type(8)))  _Float16 v8h;
typedef __attribute__((ext_vector_type(8)))  float    v8f;

union Frag16 { v16h v; v8h h[2]; };

#define B_    4096
#define D_    2048
#define PROJ_ 2730
#define NPH_  2816   // padded half of up projection (multiple of 128)
#define NUP_  5632   // 2*NPH_
#define KDN_  2752   // padded K for down GEMM (multiple of 32)

// ---------------- transpose + convert: f32 [K][Nsrc] -> f16 [Np][Kp] ----------------
__global__ __launch_bounds__(256)
void k_convT(const float* __restrict__ src, int srcLD, int colOff,
             int Kvalid, int Nvalid, h16_t* __restrict__ dst, int Kp, int Np)
{
    __shared__ float tile[32][33];
    int k0 = blockIdx.x * 32, n0 = blockIdx.y * 32;
    int tx = threadIdx.x & 31, ty = threadIdx.x >> 5;
    #pragma unroll
    for (int r = 0; r < 32; r += 8) {
        int k = k0 + ty + r, n = n0 + tx;
        float v = (k < Kvalid && n < Nvalid) ? src[(size_t)k * srcLD + colOff + n] : 0.f;
        tile[ty + r][tx] = v;
    }
    __syncthreads();
    #pragma unroll
    for (int r = 0; r < 32; r += 8) {
        int n = n0 + ty + r, k = k0 + tx;
        dst[(size_t)n * Kp + k] = (h16_t)tile[tx][ty + r];
    }
}

// ---------------- remap up-bias into padded layout ----------------
__global__ void k_ub(const float* __restrict__ up_b, float* __restrict__ ub)
{
    int n = blockIdx.x * 256 + threadIdx.x;
    if (n >= NUP_) return;
    float v = 0.f;
    if (n < NPH_) { if (n < PROJ_) v = up_b[n]; }
    else { int j = n - NPH_; if (j < PROJ_) v = up_b[PROJ_ + j]; }
    ub[n] = v;
}

// ---------------- LayerNorm (eps = D, replicating source bug) -> f16 ----------------
__global__ __launch_bounds__(256)
void k_ln(const float* __restrict__ seq, const float* __restrict__ sc,
          const float* __restrict__ bi, h16_t* __restrict__ xbf)
{
    __shared__ float red[16];
    int b = blockIdx.x, t = threadIdx.x;
    const float* row = seq + ((size_t)b << 11);
    float s = 0.f, ss = 0.f;
    for (int i = t; i < D_; i += 256) { float v = row[i]; s += v; ss += v * v; }
    for (int off = 16; off > 0; off >>= 1) { s += __shfl_down(s, off); ss += __shfl_down(ss, off); }
    int wv = t >> 5, ln = t & 31;
    if (ln == 0) { red[wv] = s; red[8 + wv] = ss; }
    __syncthreads();
    if (t == 0) {
        float ts = 0.f, tss = 0.f;
        for (int w2 = 0; w2 < 8; w2++) { ts += red[w2]; tss += red[8 + w2]; }
        red[0] = ts; red[8] = tss;
    }
    __syncthreads();
    float mean = red[0] * (1.f / 2048.f);
    float var  = red[8] * (1.f / 2048.f) - mean * mean;
    float rstd = rsqrtf(var + 2048.0f);  // EPS_LN == inp_dim
    h16_t* orow = xbf + ((size_t)b << 11);
    for (int i = t; i < D_; i += 256)
        orow[i] = (h16_t)((row[i] - mean) * rstd * sc[i] + bi[i]);
}

// ---------------- WMMA f16 GEMM: C[M][N] = A[M][K] * Bt[N][K]^T + bias + resid ----------------
// 128 threads = 4 waves (2x2), block tile 128x128x32, wave tile 64x64.
// Software pipeline: register-staged global loads for tile k+1 overlap the 16 WMMAs
// of tile k; double-buffered LDS -> one barrier per K-step.
#define BM 128
#define BN 128
#define BK 32
__global__ __launch_bounds__(128)
void k_gemm(const h16_t* __restrict__ A, const h16_t* __restrict__ Bt,
            const float* __restrict__ bias, const float* __restrict__ resid,
            float* __restrict__ C, int M, int N, int K)
{
    __shared__ h16_t As[2][BM * BK];
    __shared__ h16_t Bs[2][BN * BK];
    int m0 = blockIdx.y * BM, n0 = blockIdx.x * BN;
    int tid = threadIdx.x;
    int wave = tid >> 5, lane = tid & 31;
    int wm = (wave >> 1) * 64, wn = (wave & 1) * 64;
    int lane16 = lane & 15, hsel = lane >> 4;

    v8f acc[4][4] = {};

    // each thread stages one 32-element row of A and of B per K-step (4x b128 each)
    const v8h* gA = (const v8h*)(A + (size_t)(m0 + tid) * K);
    const v8h* gB = (const v8h*)(Bt + (size_t)(n0 + tid) * K);

    v8h ra[4], rb[4];
    #pragma unroll
    for (int i = 0; i < 4; i++) { ra[i] = gA[i]; rb[i] = gB[i]; }
    {
        v8h* sA = (v8h*)(As[0] + tid * BK);
        v8h* sB = (v8h*)(Bs[0] + tid * BK);
        #pragma unroll
        for (int i = 0; i < 4; i++) { sA[i] = ra[i]; sB[i] = rb[i]; }
    }

    int nk = K / BK;
    for (int kk = 0; kk < nk; kk++) {
        int cur = kk & 1;
        __syncthreads();
        bool more = (kk + 1) < nk;
        if (more) {
            const v8h* pA = gA + (size_t)(kk + 1) * 4;
            const v8h* pB = gB + (size_t)(kk + 1) * 4;
            #pragma unroll
            for (int i = 0; i < 4; i++) { ra[i] = pA[i]; rb[i] = pB[i]; }
            if (kk + 2 < nk) {
                __builtin_prefetch(gA + (size_t)(kk + 2) * 4, 0, 0);  // global_prefetch_b8
                __builtin_prefetch(gB + (size_t)(kk + 2) * 4, 0, 0);
            }
        }

        v16h a[4], bfr[4];
        #pragma unroll
        for (int i = 0; i < 4; i++) {
            // A frag: lanes 0-15 rows M, K 0..7 & 16..23 ; lanes 16-31 K 8..15 & 24..31
            const h16_t* p = As[cur] + (wm + i * 16 + lane16) * BK + hsel * 8;
            Frag16 fa; fa.h[0] = *(const v8h*)p; fa.h[1] = *(const v8h*)(p + 16);
            a[i] = fa.v;
            const h16_t* q = Bs[cur] + (wn + i * 16 + lane16) * BK + hsel * 8;
            Frag16 fb; fb.h[0] = *(const v8h*)q; fb.h[1] = *(const v8h*)(q + 16);
            bfr[i] = fb.v;
        }
        #pragma unroll
        for (int i = 0; i < 4; i++)
            #pragma unroll
            for (int j = 0; j < 4; j++)
                acc[i][j] = __builtin_amdgcn_wmma_f32_16x16x32_f16(
                    false, a[i], false, bfr[j], (short)0, acc[i][j], false, false);

        if (more) {
            v8h* sA = (v8h*)(As[cur ^ 1] + tid * BK);
            v8h* sB = (v8h*)(Bs[cur ^ 1] + tid * BK);
            #pragma unroll
            for (int i = 0; i < 4; i++) { sA[i] = ra[i]; sB[i] = rb[i]; }
        }
    }

    #pragma unroll
    for (int i = 0; i < 4; i++) {
        int rowbase = m0 + wm + i * 16 + hsel * 8;
        #pragma unroll
        for (int j = 0; j < 4; j++) {
            int col = n0 + wn + j * 16 + lane16;
            float bb = bias ? bias[col] : 0.f;
            #pragma unroll
            for (int v = 0; v < 8; v++) {
                int row = rowbase + v;
                size_t idx = (size_t)row * N + col;
                float val = acc[i][j][v] + bb;
                if (resid) val += resid[idx];
                C[idx] = val;
            }
        }
    }
}

// ---------------- gating + block-diagonal recurrence, writes c,n,h,m ----------------
__global__ __launch_bounds__(256)
void k_gate(const float* __restrict__ Gz, const float* __restrict__ Gi,
            const float* __restrict__ Gf, const float* __restrict__ Go,
            const float* __restrict__ Rzw, const float* __restrict__ Rzb,
            const float* __restrict__ Riw, const float* __restrict__ Rib,
            const float* __restrict__ Rfw, const float* __restrict__ Rfb,
            const float* __restrict__ Row_, const float* __restrict__ Rob,
            const float* __restrict__ h_p, const float* __restrict__ c_p,
            const float* __restrict__ n_p, const float* __restrict__ m_p,
            float* __restrict__ c_o, float* __restrict__ n_o,
            float* __restrict__ h_o, float* __restrict__ m_o)
{
    size_t idx = (size_t)blockIdx.x * 256 + threadIdx.x;
    int b = (int)(idx >> 11);
    int n = (int)(idx & 2047);
    int head = n >> 7, rem = n & 127, blk = rem >> 3, e = rem & 7;
    const float* h8 = h_p + ((size_t)b << 11) + head * 128 + blk * 8;
    int wb = blk * 64 + e;
    float az = Rzb[blk * 8 + e], ai = Rib[blk * 8 + e];
    float af = Rfb[blk * 8 + e], ao = Rob[blk * 8 + e];
    #pragma unroll
    for (int d = 0; d < 8; d++) {
        float hv = h8[d];
        az += hv * Rzw[wb + d * 8];
        ai += hv * Riw[wb + d * 8];
        af += hv * Rfw[wb + d * 8];
        ao += hv * Row_[wb + d * 8];
    }
    float it = Gi[idx] + ai, ft = Gf[idx] + af;
    float zt = Gz[idx] + az, ot = Go[idx] + ao;
    float mp = m_p[idx];
    float mt = fmaxf(ft + mp, it);
    float ie = expf(it - mt);
    float fe = expf(ft - mt + mp);
    float z  = tanhf(zt);
    float o  = 1.f / (1.f + expf(-ot));
    float ct = fe * c_p[idx] + ie * z;
    float nt = fe * n_p[idx] + ie;
    float ht = o * (ct / nt);
    c_o[idx] = ct; n_o[idx] = nt; h_o[idx] = ht; m_o[idx] = mt;
}

// ---------------- GroupNorm (stats over H and Dg per (b,g)) -> f16 ----------------
__global__ __launch_bounds__(256)
void k_gn(const float* __restrict__ hT, const float* __restrict__ gsc,
          const float* __restrict__ gbi, h16_t* __restrict__ out)
{
    __shared__ float sbuf[256], ssbuf[256];
    int b = blockIdx.x, t = threadIdx.x;
    int g = t >> 4, u = t & 15;         // g: group 0..15, u: head index 0..15
    const float* row = hT + ((size_t)b << 11);
    float s = 0.f, ss = 0.f;
    #pragma unroll
    for (int d = 0; d < 8; d++) {
        float v = row[u * 128 + g * 8 + d];
        s += v; ss += v * v;
    }
    sbuf[t] = s; ssbuf[t] = ss;
    for (int off = 8; off > 0; off >>= 1) {
        __syncthreads();
        if (u < off) { sbuf[t] += sbuf[t + off]; ssbuf[t] += ssbuf[t + off]; }
    }
    __syncthreads();
    float mean = sbuf[g << 4] * (1.f / 128.f);
    float var  = ssbuf[g << 4] * (1.f / 128.f) - mean * mean;
    float rstd = rsqrtf(var + 1e-6f);
    h16_t* orow = out + ((size_t)b << 11);
    #pragma unroll
    for (int d = 0; d < 8; d++) {
        int c = g * 8 + d;
        int idx = u * 128 + c;
        orow[idx] = (h16_t)((row[idx] - mean) * rstd * gsc[c] + gbi[c]);
    }
}

// ---------------- GLU: h2 = out1 + gelu(out2) (tanh approx), f16, zero-padded ----------------
__global__ void k_glu(const float* __restrict__ U, h16_t* __restrict__ h2)
{
    size_t idx = (size_t)blockIdx.x * 256 + threadIdx.x;
    int b = (int)(idx / KDN_);
    int j = (int)(idx - (size_t)b * KDN_);
    float v = 0.f;
    if (j < PROJ_) {
        float a = U[(size_t)b * NUP_ + j];
        float x = U[(size_t)b * NUP_ + NPH_ + j];
        float g = 0.5f * x * (1.f + tanhf(0.7978845608028654f * (x + 0.044715f * x * x * x)));
        v = a + g;
    }
    h2[idx] = (h16_t)v;
}

// ---------------- host launch ----------------
extern "C" void kernel_launch(void* const* d_in, const int* in_sizes, int n_in,
                              void* d_out, int out_size, void* d_ws, size_t ws_size,
                              hipStream_t stream)
{
    (void)in_sizes; (void)n_in; (void)out_size; (void)ws_size;

    const float* seq   = (const float*)d_in[0];
    const float* c_tm1 = (const float*)d_in[1];
    const float* n_tm1 = (const float*)d_in[2];
    const float* h_tm1 = (const float*)d_in[3];
    const float* m_tm1 = (const float*)d_in[4];
    const float* ln_s  = (const float*)d_in[5];
    const float* ln_b  = (const float*)d_in[6];
    const float* Wg[4]  = { (const float*)d_in[7],  (const float*)d_in[9],
                            (const float*)d_in[11], (const float*)d_in[13] };   // z,i,f,o
    const float* bWg[4] = { (const float*)d_in[8],  (const float*)d_in[10],
                            (const float*)d_in[12], (const float*)d_in[14] };
    const float* Rzw = (const float*)d_in[15]; const float* Rzb = (const float*)d_in[16];
    const float* Riw = (const float*)d_in[17]; const float* Rib = (const float*)d_in[18];
    const float* Rfw = (const float*)d_in[19]; const float* Rfb = (const float*)d_in[20];
    const float* Row_= (const float*)d_in[21]; const float* Rob = (const float*)d_in[22];
    const float* gn_s = (const float*)d_in[23];
    const float* gn_b = (const float*)d_in[24];
    const float* up_w = (const float*)d_in[25];
    const float* up_b = (const float*)d_in[26];
    const float* dn_w = (const float*)d_in[27];
    const float* dn_b = (const float*)d_in[28];

    // ---- workspace carve-up ----
    char* w = (char*)d_ws;
    h16_t* xbf = (h16_t*)w;            w += (size_t)B_ * D_ * 2;
    h16_t* Wt[4];
    for (int g = 0; g < 4; g++) { Wt[g] = (h16_t*)w; w += (size_t)D_ * D_ * 2; }
    h16_t* upt = (h16_t*)w;            w += (size_t)NUP_ * D_ * 2;
    h16_t* dnt = (h16_t*)w;            w += (size_t)D_ * KDN_ * 2;
    float* ub  = (float*)w;            w += (size_t)NUP_ * 4;       // 22528B, 256-aligned
    h16_t* gnbf= (h16_t*)w;            w += (size_t)B_ * D_ * 2;
    char*  big = w;                                                    // 134,217,728 B region
    float* G[4];
    for (int g = 0; g < 4; g++) G[g] = (float*)(big + (size_t)g * B_ * D_ * 4);
    float* U   = (float*)big;                                          // reuses G (consumed first)
    h16_t* h2  = (h16_t*)(big + (size_t)B_ * NUP_ * 4);                // after U

    // ---- d_out sections: out, c, n, h, m ----
    float* out0 = (float*)d_out;
    float* c_o = out0 + (size_t)B_ * D_;
    float* n_o = c_o + (size_t)B_ * D_;
    float* h_o = n_o + (size_t)B_ * D_;
    float* m_o = h_o + (size_t)B_ * D_;

    // 1) weight convert+transpose to f16 [N][K]
    for (int g = 0; g < 4; g++)
        k_convT<<<dim3(64, 64), 256, 0, stream>>>(Wg[g], D_, 0, D_, D_, Wt[g], D_, D_);
    k_convT<<<dim3(64, 88), 256, 0, stream>>>(up_w, 2 * PROJ_, 0,     D_, PROJ_, upt,                    D_,   NPH_);
    k_convT<<<dim3(64, 88), 256, 0, stream>>>(up_w, 2 * PROJ_, PROJ_, D_, PROJ_, upt + (size_t)NPH_ * D_, D_,   NPH_);
    k_convT<<<dim3(86, 64), 256, 0, stream>>>(dn_w, D_, 0, PROJ_, D_, dnt, KDN_, D_);
    k_ub<<<dim3((NUP_ + 255) / 256), 256, 0, stream>>>(up_b, ub);

    // 2) LayerNorm -> f16 activations
    k_ln<<<dim3(B_), 256, 0, stream>>>(seq, ln_s, ln_b, xbf);

    // 3) gate GEMMs (z,i,f,o): G = x @ W + b
    for (int g = 0; g < 4; g++)
        k_gemm<<<dim3(D_ / BN, B_ / BM), 128, 0, stream>>>(xbf, Wt[g], bWg[g], nullptr,
                                                           G[g], B_, D_, D_);

    // 4) gating + recurrence -> c,n,h,m in d_out
    k_gate<<<dim3((B_ * D_) / 256), 256, 0, stream>>>(G[0], G[1], G[2], G[3],
        Rzw, Rzb, Riw, Rib, Rfw, Rfb, Row_, Rob,
        h_tm1, c_tm1, n_tm1, m_tm1, c_o, n_o, h_o, m_o);

    // 5) GroupNorm(h_t) -> f16
    k_gn<<<dim3(B_), 256, 0, stream>>>(h_o, gn_s, gn_b, gnbf);

    // 6) up GEMM (both halves in padded layout)
    k_gemm<<<dim3(NUP_ / BN, B_ / BM), 128, 0, stream>>>(gnbf, upt, ub, nullptr,
                                                         U, B_, NUP_, D_);

    // 7) GLU combine -> f16 (zero-padded K for down GEMM)
    k_glu<<<dim3((B_ * KDN_) / 256), 256, 0, stream>>>(U, h2);

    // 8) down GEMM + bias + residual -> d_out[0]
    k_gemm<<<dim3(D_ / BN, B_ / BM), 128, 0, stream>>>(h2, dnt, dn_b, seq,
                                                       out0, B_, D_, KDN_);
}
